// BaseMDRNN_78623671320579
// MI455X (gfx1250) — compile-verified, matrix operand order checked by reading, not run on hardware
//
#include <hip/hip_runtime.h>
#include <hip/hip_bf16.h>
#include <math.h>

// MDRNN on MI455X (gfx1250, wave32, WMMA 16x16x32 bf16, f32 accumulate).
//
// Phase 0a: zero per-cell ready flags in d_ws (deterministic across replays).
// Phase 0b: pack Wx/Wu/Wl (f32) into bf16 WMMA B-fragment layout in d_ws.
// Phase 1 : x_proj = x @ Wx + b  -> written straight into d_out [B,H,W,U] f32.
// Phase 2 : ONE persistent dataflow kernel. Block i owns grid row i and
//           marches j = 0..W-1. a[i][j-1] is kept in LDS (no global trip);
//           a[i-1][j] readiness is signaled via agent-scope flags. All
//           dependencies point at lower block IDs -> schedulable in order,
//           and 256 blocks are trivially co-resident on MI455X.

typedef __attribute__((ext_vector_type(16))) __bf16 v16bf;
typedef __attribute__((ext_vector_type(8)))  float  v8f;

#define MD_B 8
#define MD_H 256
#define MD_W 256
#define MD_U 64
#define LDS_PITCH 68                  // 64 + 4 pad: spreads fragment reads over banks
#define MD_FLAGS_OFF (32 * 1024)      // flags live after the 24 KB weight pack

// 16-bit A/B fragment K index for element e (0..15) given lane half (0/1).
__device__ __forceinline__ int kmap(int e, int half) {
    return ((e & 8) << 1) + (e & 7) + half * 8;
}

__device__ __forceinline__ float fast_tanh(float x) {
#if __has_builtin(__builtin_amdgcn_tanhf)
    return __builtin_amdgcn_tanhf(x);
#elif __has_builtin(__builtin_amdgcn_tanh_f32)
    return __builtin_amdgcn_tanh_f32(x);
#else
    return tanhf(x);
#endif
}

// Build a 16-bit A fragment from a 64-float row segment: element e maps to
// k = kmap(e, half), two contiguous 8-float runs -> 4x float4 loads.
__device__ __forceinline__ v16bf pack_a(const float* __restrict__ p, int half) {
    const float* q = p + half * 8;
    float4 r0 = *(const float4*)(q + 0);
    float4 r1 = *(const float4*)(q + 4);
    float4 r2 = *(const float4*)(q + 16);
    float4 r3 = *(const float4*)(q + 20);
    v16bf v;
    v[0]  = (__bf16)r0.x; v[1]  = (__bf16)r0.y; v[2]  = (__bf16)r0.z; v[3]  = (__bf16)r0.w;
    v[4]  = (__bf16)r1.x; v[5]  = (__bf16)r1.y; v[6]  = (__bf16)r1.z; v[7]  = (__bf16)r1.w;
    v[8]  = (__bf16)r2.x; v[9]  = (__bf16)r2.y; v[10] = (__bf16)r2.z; v[11] = (__bf16)r2.w;
    v[12] = (__bf16)r3.x; v[13] = (__bf16)r3.y; v[14] = (__bf16)r3.z; v[15] = (__bf16)r3.w;
    return v;
}

// Packed-weight fragment: frag = ((mat*2 + ks)*4 + nt), 16 bf16 per lane.
__device__ __forceinline__ v16bf load_wfrag(const __bf16* __restrict__ wpack,
                                            int mat, int ks, int nt, int lane) {
    const __bf16* p = wpack + (((long)((mat * 2 + ks) * 4 + nt) * 32) + lane) * 16;
    return *(const v16bf*)p;
}

// ---------------------------------------------------------------------------
// Phase 0a: zero the H*W ready flags.
// ---------------------------------------------------------------------------
__global__ void mdrnn_zero_flags(int* __restrict__ flags) {
    const int t = blockIdx.x * blockDim.x + threadIdx.x;
    if (t < MD_H * MD_W) flags[t] = 0;
}

// ---------------------------------------------------------------------------
// Phase 0b: pack 3 weight matrices into B-fragment layout (768 threads).
// ---------------------------------------------------------------------------
__global__ void mdrnn_pack_weights(const float* __restrict__ Wx,
                                   const float* __restrict__ Wu,
                                   const float* __restrict__ Wl,
                                   __bf16* __restrict__ wpack) {
    const int t = blockIdx.x * blockDim.x + threadIdx.x;
    if (t >= 24 * 32) return;
    const int lane = t & 31;
    const int frag = t >> 5;          // 0..23
    const int nt   = frag & 3;
    const int ks   = (frag >> 2) & 1;
    const int mat  = frag >> 3;       // 0=Wx, 1=Wu, 2=Wl
    const float* Wm = (mat == 0) ? Wx : (mat == 1) ? Wu : Wl;
    const int row  = lane & 15;
    const int half = lane >> 4;
    const int n    = nt * 16 + row;
    __bf16* dst = wpack + ((long)frag * 32 + lane) * 16;
    #pragma unroll
    for (int e = 0; e < 16; ++e) {
        dst[e] = (__bf16)Wm[(kmap(e, half) + 32 * ks) * MD_U + n];
    }
}

// ---------------------------------------------------------------------------
// Phase 1: dense GEMM  out[m][u] = sum_k x[m][k]*Wx[k][u] + b[u], K=N=64.
// One wave per 16-row M tile; 8 waves per block; grid exact.
// ---------------------------------------------------------------------------
__global__ void mdrnn_xproj_kernel(const float* __restrict__ x,
                                   const __bf16* __restrict__ wpack,
                                   const float* __restrict__ bias,
                                   float* __restrict__ out) {
    const int lane = threadIdx.x & 31;
    const int wave = threadIdx.x >> 5;
    const int row  = lane & 15;
    const int half = lane >> 4;
    const long m0 = ((long)blockIdx.x * 8 + wave) * 16;

    const float* xrow = x + (m0 + row) * MD_U;
    const v16bf a0 = pack_a(xrow, half);
    const v16bf a1 = pack_a(xrow + 32, half);

    #pragma unroll
    for (int nt = 0; nt < 4; ++nt) {
        const int n = nt * 16 + row;
        const v16bf b0 = load_wfrag(wpack, 0, 0, nt, lane);
        const v16bf b1 = load_wfrag(wpack, 0, 1, nt, lane);
        v8f acc = {};
        acc = __builtin_amdgcn_wmma_f32_16x16x32_bf16(false, a0, false, b0,
                                                      (short)0, acc, false, false);
        acc = __builtin_amdgcn_wmma_f32_16x16x32_bf16(false, a1, false, b1,
                                                      (short)0, acc, false, false);
        const float bv = bias[n];
        #pragma unroll
        for (int r = 0; r < 8; ++r) {
            out[(m0 + r + 8 * half) * MD_U + n] = acc[r] + bv;
        }
    }
}

// ---------------------------------------------------------------------------
// Phase 2: persistent row-dataflow scan. Block i = grid row i, 4 waves,
// wave nt owns output columns [16*nt, 16*nt+16) of each cell.
// State tensor a == d_out, [B,H,W,U] f32 (x_proj before a cell is processed,
// activation after). 128 MB -> L2-resident; flag sync is L2-latency, not
// kernel-launch latency.
// ---------------------------------------------------------------------------
__global__ void mdrnn_scan_rows(const __bf16* __restrict__ wpack,
                                float* __restrict__ a,
                                int* __restrict__ flags) {
    const int i    = (int)blockIdx.x;      // grid row
    const int lane = threadIdx.x & 31;
    const int nt   = threadIdx.x >> 5;
    const int row  = lane & 15;
    const int half = lane >> 4;
    const int n    = nt * 16 + row;
    const long sB  = (long)MD_H * MD_W * MD_U;

    __shared__ float lds_prev[MD_B * LDS_PITCH];   // a[i][j-1], bank-padded

    // Recurrent weight fragments stay in VGPRs for the whole row.
    const v16bf bu0 = load_wfrag(wpack, 1, 0, nt, lane);
    const v16bf bu1 = load_wfrag(wpack, 1, 1, nt, lane);
    const v16bf bl0 = load_wfrag(wpack, 2, 0, nt, lane);
    const v16bf bl1 = load_wfrag(wpack, 2, 1, nt, lane);

    const int* upflags = flags + (long)(i - 1) * MD_W;   // only read when i > 0

    for (int j = 0; j < MD_W; ++j) {
        // ---- A_left fragments from LDS (cell computed last iteration) ----
        v16bf al0 = {}, al1 = {};
        if (j > 0 && row < MD_B) {
            const float* p = lds_prev + row * LDS_PITCH;
            al0 = pack_a(p, half);
            al1 = pack_a(p + 32, half);
        }

        // ---- wait for up-neighbor cell (i-1, j) ----
        if (i > 0 && threadIdx.x == 0) {
            while (__hip_atomic_load(&upflags[j], __ATOMIC_RELAXED,
                                     __HIP_MEMORY_SCOPE_AGENT) == 0) {
                __builtin_amdgcn_s_sleep(1);
            }
        }
        __syncthreads();   // flag observed; all LDS reads of a_left done
        __builtin_amdgcn_fence(__ATOMIC_ACQUIRE, "agent");

        const long base_ij = ((long)i * MD_W + j) * MD_U;

        // ---- A_up fragments from global (row i-1 written by block i-1) ----
        v16bf au0 = {}, au1 = {};
        if (i > 0 && row < MD_B) {
            const float* p = a + (long)row * sB + base_ij - (long)MD_W * MD_U;
            au0 = pack_a(p, half);
            au1 = pack_a(p + 32, half);
        }

        // ---- C = x_proj fragment at (i,j) ----
        v8f acc;
        #pragma unroll
        for (int r = 0; r < 8; ++r) {
            const int mb = r + 8 * half;
            acc[r] = (mb < MD_B) ? a[(long)mb * sB + base_ij + n] : 0.f;
        }

        acc = __builtin_amdgcn_wmma_f32_16x16x32_bf16(false, au0, false, bu0,
                                                      (short)0, acc, false, false);
        acc = __builtin_amdgcn_wmma_f32_16x16x32_bf16(false, au1, false, bu1,
                                                      (short)0, acc, false, false);
        acc = __builtin_amdgcn_wmma_f32_16x16x32_bf16(false, al0, false, bl0,
                                                      (short)0, acc, false, false);
        acc = __builtin_amdgcn_wmma_f32_16x16x32_bf16(false, al1, false, bl1,
                                                      (short)0, acc, false, false);

        #pragma unroll
        for (int r = 0; r < 8; ++r) acc[r] = fast_tanh(acc[r]);

        // ---- stage for next A_left + commit output (valid half only) ----
        if (half == 0) {
            #pragma unroll
            for (int r = 0; r < 8; ++r) {
                lds_prev[r * LDS_PITCH + n] = acc[r];
                a[(long)r * sB + base_ij + n] = acc[r];
            }
        }
        // Flush this wave's global stores to agent scope, join, then signal.
        __builtin_amdgcn_fence(__ATOMIC_RELEASE, "agent");
        __syncthreads();   // also makes LDS writes visible for next iteration
        if (threadIdx.x == 0) {
            __hip_atomic_store(&flags[(long)i * MD_W + j], 1, __ATOMIC_RELAXED,
                               __HIP_MEMORY_SCOPE_AGENT);
        }
    }
}

extern "C" void kernel_launch(void* const* d_in, const int* in_sizes, int n_in,
                              void* d_out, int out_size, void* d_ws, size_t ws_size,
                              hipStream_t stream) {
    (void)in_sizes; (void)n_in; (void)out_size; (void)ws_size;

    const float* x  = (const float*)d_in[0];   // [B,H,W,D]
    const float* Wx = (const float*)d_in[1];   // [D,U]
    const float* Wu = (const float*)d_in[2];   // [U,U]
    const float* Wl = (const float*)d_in[3];   // [U,U]
    const float* b  = (const float*)d_in[4];   // [U]
    float*  out   = (float*)d_out;             // [B,H,W,U]
    __bf16* wpack = (__bf16*)d_ws;             // 24 KB packed weight fragments
    int*    flags = (int*)((char*)d_ws + MD_FLAGS_OFF);   // 256 KB ready flags

    // Phase 0a: zero flags (must happen every call -> deterministic replays).
    mdrnn_zero_flags<<<dim3((MD_H * MD_W + 255) / 256), dim3(256), 0, stream>>>(flags);

    // Phase 0b: pack weights.
    mdrnn_pack_weights<<<dim3(3), dim3(256), 0, stream>>>(Wx, Wu, Wl, wpack);

    // Phase 1: 524288 rows -> 32768 M tiles -> 4096 blocks of 8 waves (exact).
    const int m_tiles = (MD_B * MD_H * MD_W) / 16;
    mdrnn_xproj_kernel<<<dim3(m_tiles / 8), dim3(256), 0, stream>>>(x, wpack, b, out);

    // Phase 2: one persistent dataflow kernel, block per grid row.
    mdrnn_scan_rows<<<dim3(MD_H), dim3(128), 0, stream>>>(wpack, out, flags);
}
